// MultiHeadAttentionLayer_28939489641157
// MI455X (gfx1250) — compile-verified, matrix operand order checked by reading, not run on hardware
//
#include <hip/hip_runtime.h>
#include <hip/hip_bf16.h>
#include <cstdint>

// MHA for MI455X (gfx1250, wave32).
// Roofline: ~17 GFLOP vs ~268MB of mandatory mask reads -> HBM-bound on the
// [B,H,N,N] f32 masks (~11.5us floor at 23.3 TB/s). Strategy:
//  - all GEMMs in v_wmma_f32_16x16x32_f16 (f16 in, f32 accumulate)
//  - masks streamed exactly once, non-temporal (preserve L2 for the 16MB of
//    reused f16 Q/K/V/attn intermediates; L2 = 192MB)
//  - K/V tiles staged with GLOBAL_LOAD_ASYNC_TO_LDS_B128 (ASYNCcnt) when the
//    toolchain exposes the builtin; synchronous fallback otherwise.

#define B_ 4
#define N_ 1024
#define E_ 512
#define H_ 8
#define D_ 64

typedef __attribute__((ext_vector_type(16))) _Float16 v16h;
typedef __attribute__((ext_vector_type(8)))  float    v8f;
typedef __attribute__((ext_vector_type(4)))  uint32_t v4u;
typedef __attribute__((ext_vector_type(4)))  int      v4i;

#define AS1 __attribute__((address_space(1)))
#define AS3 __attribute__((address_space(3)))

#if defined(__has_builtin)
#  if __has_builtin(__builtin_amdgcn_global_load_async_to_lds_b128)
#    define USE_ASYNC_LDS 1
#  endif
#endif
#ifndef USE_ASYNC_LDS
#  define USE_ASYNC_LDS 0
#endif

__device__ __forceinline__ void wait_asynccnt0() {
#if defined(__has_builtin) && __has_builtin(__builtin_amdgcn_s_wait_asynccnt)
  __builtin_amdgcn_s_wait_asynccnt(0);
#else
  asm volatile("s_wait_asynccnt 0x0" ::: "memory");
#endif
}

// ---------------------------------------------------------------------------
// WMMA helpers
// ---------------------------------------------------------------------------
__device__ __forceinline__ v8f wmma32(v16h a, v16h b, v8f c) {
  // 8 args: (neg_a, A, neg_b, B, c_mod, C, reuse_a, reuse_b)
  return __builtin_amdgcn_wmma_f32_16x16x32_f16(
      false, a, false, b, (short)0, c, false, false);
}

// Load a 16x32 f16 fragment (A layout; same striping used for B since both
// operand tiles are stored row-major in the contraction dim). ISA 7.12.2:
//   lanes 0-15 : row = lane,     K groups {0..7, 16..23}
//   lanes16-31 : row = lane-16,  K groups {8..15, 24..31}
// Each half-wave's K group is 8 contiguous halfs -> two b128 loads per lane.
__device__ __forceinline__ v16h load_frag16(const _Float16* base, int ldh) {
  const int lane = threadIdx.x & 31;
  const int row  = lane & 15;
  const int hi   = lane >> 4;
  union { v16h v; v4u q[2]; } f;
  const _Float16* rp = base + row * ldh + hi * 8;
  f.q[0] = *(const v4u*)(rp);
  f.q[1] = *(const v4u*)(rp + 16);
  return f.v;
}

// Row reductions over 16-lane halves (C/D layout: lanes 0-15 and 16-31 hold
// independent row groups; xor masks < 16 stay within a half).
__device__ __forceinline__ float rowmax16(float x) {
  x = fmaxf(x, __shfl_xor(x, 1, 32));
  x = fmaxf(x, __shfl_xor(x, 2, 32));
  x = fmaxf(x, __shfl_xor(x, 4, 32));
  x = fmaxf(x, __shfl_xor(x, 8, 32));
  return x;
}
__device__ __forceinline__ float rowsum16(float x) {
  x += __shfl_xor(x, 1, 32);
  x += __shfl_xor(x, 2, 32);
  x += __shfl_xor(x, 4, 32);
  x += __shfl_xor(x, 8, 32);
  return x;
}

// ---------------------------------------------------------------------------
// f32 -> f16 conversion
// ---------------------------------------------------------------------------
__global__ void f32_to_f16_kernel(const float* __restrict__ src,
                                  _Float16* __restrict__ dst, int n) {
  int i = blockIdx.x * blockDim.x + threadIdx.x;
  if (i < n) dst[i] = (_Float16)src[i];
}

// ---------------------------------------------------------------------------
// GEMM: out[r, c] = sum_k A[r, k] * W[c, k] + bias[c]   (torch Linear)
// A: [4096, 512] f16, W: [512, 512] f16. 8 waves, each a 32x32 tile;
// block tile 128x64. Operand frags loaded straight from global (L2-resident).
// ---------------------------------------------------------------------------
template <typename OutT>
__global__ void __launch_bounds__(256) gemm_xwT_kernel(
    const _Float16* __restrict__ A, const _Float16* __restrict__ W,
    const float* __restrict__ bias, OutT* __restrict__ out) {
  const int rowBase = blockIdx.x * 128;
  const int colBase = blockIdx.y * 64;
  const int wid = threadIdx.x >> 5;
  const int r0 = rowBase + (wid & 3) * 32;
  const int c0 = colBase + (wid >> 2) * 32;

  v8f acc[2][2] = {};
  for (int k0 = 0; k0 < E_; k0 += 32) {
    v16h a0 = load_frag16(A + (size_t)r0 * E_ + k0, E_);
    v16h a1 = load_frag16(A + (size_t)(r0 + 16) * E_ + k0, E_);
    v16h b0 = load_frag16(W + (size_t)c0 * E_ + k0, E_);
    v16h b1 = load_frag16(W + (size_t)(c0 + 16) * E_ + k0, E_);
    acc[0][0] = wmma32(a0, b0, acc[0][0]);
    acc[0][1] = wmma32(a0, b1, acc[0][1]);
    acc[1][0] = wmma32(a1, b0, acc[1][0]);
    acc[1][1] = wmma32(a1, b1, acc[1][1]);
  }

  const int lane = threadIdx.x & 31;
  const int col = lane & 15;
  const int hi  = lane >> 4;
#pragma unroll
  for (int i = 0; i < 2; ++i) {
#pragma unroll
    for (int j = 0; j < 2; ++j) {
      const int c = c0 + j * 16 + col;
      const float bv = bias[c];
#pragma unroll
      for (int v = 0; v < 8; ++v) {
        const int r = r0 + i * 16 + v + 8 * hi;
        out[(size_t)r * E_ + c] = (OutT)(acc[i][j][v] + bv);
      }
    }
  }
}

// ---------------------------------------------------------------------------
// Fused attention (flash-style; masks streamed once, non-temporal).
// Block = 8 waves; each wave owns 16 query rows of one (b,h); key tiles of 64.
// Q/K/V f16 laid out [b, n, h, d] so head slices are contiguous.
// ---------------------------------------------------------------------------
__global__ void __launch_bounds__(256) attn_kernel(
    const _Float16* __restrict__ qb, const _Float16* __restrict__ kb,
    const _Float16* __restrict__ vb, const float* __restrict__ rnodes,
    const float* __restrict__ aadd, const float* __restrict__ amul,
    _Float16* __restrict__ ob) {
  __shared__ alignas(16) _Float16 Ks[64 * 72];     // [key][d] -> B frags (QK^T)
  __shared__ alignas(16) _Float16 Vt[64 * 72];     // [d][key] -> B frags (P@V)
  __shared__ alignas(16) _Float16 Ps[8][16 * 72];  // per-wave probs C->A restripe
#if USE_ASYNC_LDS
  __shared__ alignas(16) _Float16 Vraw[64 * 64];   // async landing pad for V
#endif

  const int b = blockIdx.z;
  const int h = blockIdx.y;
  const int wid  = threadIdx.x >> 5;
  const int lane = threadIdx.x & 31;
  const int col  = lane & 15;
  const int hi   = lane >> 4;
  const int qbase = blockIdx.x * 128 + wid * 16;

  // Q fragments live in registers for the whole pass (K = D = 64 -> 2 frags).
  const _Float16* qp = qb + ((size_t)(b * N_) + qbase) * E_ + h * D_;
  const v16h aq0 = load_frag16(qp, E_);
  const v16h aq1 = load_frag16(qp + 32, E_);

  float rn[8];
#pragma unroll
  for (int v = 0; v < 8; ++v)
    rn[v] = rnodes[b * N_ + qbase + v + 8 * hi] * 0.125f;  // 1/sqrt(64)

  const size_t mbase = ((size_t)(b * H_ + h)) * N_ * N_;

  v8f acc[4] = {};
  float l[8], m[8];
#pragma unroll
  for (int v = 0; v < 8; ++v) { l[v] = 0.0f; m[v] = -1e30f; }

  for (int kt = 0; kt < N_; kt += 64) {
    const _Float16* ksrc = kb + ((size_t)(b * N_) + kt) * E_ + h * D_;
    const _Float16* vsrc = vb + ((size_t)(b * N_) + kt) * E_ + h * D_;

    // Prefetch next tile's mask rows (dominant HBM stream).
    if (kt + 64 < N_) {
#pragma unroll
      for (int v = 0; v < 8; ++v) {
        const size_t off =
            mbase + (size_t)(qbase + v + 8 * hi) * N_ + (kt + 64) + col;
        __builtin_prefetch(aadd + off, 0, 0);
        __builtin_prefetch(amul + off, 0, 0);
      }
    }

#if USE_ASYNC_LDS
    // --- async b128 staging of K (padded rows) and raw V, ASYNCcnt-tracked ---
    for (int idx = threadIdx.x; idx < 64 * 8; idx += 256) {
      const int key = idx >> 3, c = (idx & 7) * 8;  // 8 halfs = 16B per chunk
      __builtin_amdgcn_global_load_async_to_lds_b128(
          (v4i*)(ksrc + (size_t)key * E_ + c),
          (AS3 v4i*)&Ks[key * 72 + c], 0, 0);
      __builtin_amdgcn_global_load_async_to_lds_b128(
          (v4i*)(vsrc + (size_t)key * E_ + c),
          (AS3 v4i*)&Vraw[key * 64 + c], 0, 0);
    }
    wait_asynccnt0();
    __syncthreads();
    // LDS->LDS transpose of V into [d][key]
    for (int idx = threadIdx.x; idx < 64 * 64; idx += 256) {
      const int key = idx >> 6, d = idx & 63;
      Vt[d * 72 + key] = Vraw[key * 64 + d];
    }
#else
    // --- synchronous staging fallback ---
    for (int idx = threadIdx.x; idx < 64 * 32; idx += 256) {
      const int key = idx >> 5, dp = (idx & 31) * 2;
      *(uint32_t*)&Ks[key * 72 + dp] =
          *(const uint32_t*)(ksrc + (size_t)key * E_ + dp);
    }
    for (int idx = threadIdx.x; idx < 64 * 64; idx += 256) {
      const int key = idx >> 6, d = idx & 63;
      Vt[d * 72 + key] = vsrc[(size_t)key * E_ + d];
    }
#endif
    __syncthreads();

    // --- scores: S = (Q K^T) * (rn/8) + attn_add (NT stream) ---
    v8f s[4];
#pragma unroll
    for (int nf = 0; nf < 4; ++nf) {
      v8f sc = {};
      sc = wmma32(aq0, load_frag16(&Ks[nf * 16 * 72], 72), sc);
      sc = wmma32(aq1, load_frag16(&Ks[nf * 16 * 72 + 32], 72), sc);
      const float* ap = aadd + mbase + (size_t)(kt + nf * 16 + col);
#pragma unroll
      for (int v = 0; v < 8; ++v) {
        const int q = qbase + v + 8 * hi;
        const float av = __builtin_nontemporal_load(&ap[(size_t)q * N_]);
        sc[v] = sc[v] * rn[v] + av;
      }
      s[nf] = sc;
    }

    // --- online softmax; fold attn_mult post-exp ---
    float alpha[8];
#pragma unroll
    for (int v = 0; v < 8; ++v) {
      float x = fmaxf(fmaxf(s[0][v], s[1][v]), fmaxf(s[2][v], s[3][v]));
      x = rowmax16(x);
      const float nm = fmaxf(m[v], x);
      alpha[v] = __expf(m[v] - nm);
      m[v] = nm;
    }
    float psum[8];
#pragma unroll
    for (int v = 0; v < 8; ++v) psum[v] = 0.0f;
#pragma unroll
    for (int nf = 0; nf < 4; ++nf) {
      const float* mp = amul + mbase + (size_t)(kt + nf * 16 + col);
#pragma unroll
      for (int v = 0; v < 8; ++v) {
        const int q = qbase + v + 8 * hi;
        float p = __expf(s[nf][v] - m[v]);
        psum[v] += p;  // normalizer excludes the multiplicative mask
        p *= __builtin_nontemporal_load(&mp[(size_t)q * N_]);
        Ps[wid][(v + 8 * hi) * 72 + nf * 16 + col] = (_Float16)p;
      }
    }
#pragma unroll
    for (int v = 0; v < 8; ++v) l[v] = l[v] * alpha[v] + rowsum16(psum[v]);

    // --- O = O*alpha + P @ V (probs re-striped via per-wave LDS scratch) ---
    const v16h ap0 = load_frag16(&Ps[wid][0], 72);
    const v16h ap1 = load_frag16(&Ps[wid][32], 72);
#pragma unroll
    for (int df = 0; df < 4; ++df) {
#pragma unroll
      for (int v = 0; v < 8; ++v) acc[df][v] *= alpha[v];
      acc[df] = wmma32(ap0, load_frag16(&Vt[df * 16 * 72], 72), acc[df]);
      acc[df] = wmma32(ap1, load_frag16(&Vt[df * 16 * 72 + 32], 72), acc[df]);
    }
    __syncthreads();
  }

  // --- normalize and store [b, q, h, d] (matches transpose+reshape) ---
  _Float16* op = ob + ((size_t)(b * N_) + qbase) * E_ + h * D_;
#pragma unroll
  for (int v = 0; v < 8; ++v) {
    const float inv = 1.0f / l[v];
    const int r = v + 8 * hi;
#pragma unroll
    for (int df = 0; df < 4; ++df)
      op[(size_t)r * E_ + df * 16 + col] = (_Float16)(acc[df][v] * inv);
  }
}

// ---------------------------------------------------------------------------
// Launch
// ---------------------------------------------------------------------------
extern "C" void kernel_launch(void* const* d_in, const int* in_sizes, int n_in,
                              void* d_out, int out_size, void* d_ws, size_t ws_size,
                              hipStream_t stream) {
  (void)in_sizes; (void)n_in; (void)out_size; (void)ws_size;
  const float* X    = (const float*)d_in[0];
  const float* rno  = (const float*)d_in[1];
  const float* aadd = (const float*)d_in[2];
  const float* amul = (const float*)d_in[3];
  const float* Wq   = (const float*)d_in[4];
  const float* bq   = (const float*)d_in[5];
  const float* Wk   = (const float*)d_in[6];
  const float* bk   = (const float*)d_in[7];
  const float* Wv   = (const float*)d_in[8];
  const float* bv   = (const float*)d_in[9];
  const float* Wo   = (const float*)d_in[10];
  const float* bo   = (const float*)d_in[11];
  float* out = (float*)d_out;

  // Workspace layout (f16): ~23 MB total.
  const int BNE = B_ * N_ * E_;   // 2,097,152
  const int EE  = E_ * E_;        //   262,144
  _Float16* x16  = (_Float16*)d_ws;
  _Float16* wq16 = x16 + BNE;
  _Float16* wk16 = wq16 + EE;
  _Float16* wv16 = wk16 + EE;
  _Float16* wo16 = wv16 + EE;
  _Float16* q16  = wo16 + EE;
  _Float16* k16  = q16 + BNE;
  _Float16* v16  = k16 + BNE;
  _Float16* a16  = v16 + BNE;

  const dim3 blk(256);
  f32_to_f16_kernel<<<(BNE + 255) / 256, blk, 0, stream>>>(X, x16, BNE);
  f32_to_f16_kernel<<<(EE + 255) / 256, blk, 0, stream>>>(Wq, wq16, EE);
  f32_to_f16_kernel<<<(EE + 255) / 256, blk, 0, stream>>>(Wk, wk16, EE);
  f32_to_f16_kernel<<<(EE + 255) / 256, blk, 0, stream>>>(Wv, wv16, EE);
  f32_to_f16_kernel<<<(EE + 255) / 256, blk, 0, stream>>>(Wo, wo16, EE);

  const dim3 ggrid(B_ * N_ / 128, E_ / 64);   // 32 x 8
  gemm_xwT_kernel<_Float16><<<ggrid, blk, 0, stream>>>(x16, wq16, bq, q16);
  gemm_xwT_kernel<_Float16><<<ggrid, blk, 0, stream>>>(x16, wk16, bk, k16);
  gemm_xwT_kernel<_Float16><<<ggrid, blk, 0, stream>>>(x16, wv16, bv, v16);

  attn_kernel<<<dim3(N_ / 128, H_, B_), blk, 0, stream>>>(
      q16, k16, v16, rno, aadd, amul, a16);

  gemm_xwT_kernel<float><<<ggrid, blk, 0, stream>>>(a16, wo16, bo, out);
}
